// TypedConstraintGraphStateEncoder_51823075393576
// MI455X (gfx1250) — compile-verified, matrix operand order checked by reading, not run on hardware
//
#include <hip/hip_runtime.h>
#include <hip/hip_bf16.h>

#define DEV_INLINE __device__ __forceinline__

typedef __attribute__((ext_vector_type(16))) _Float16 v16h;
typedef __attribute__((ext_vector_type(8)))  _Float16 v8h;
typedef __attribute__((ext_vector_type(8)))  float    v8f;

static constexpr int NN = 8192;
static constexpr int EE = 262144;
static constexpr int HH = 128;

// Load one lane's 16-half fragment for a 16x32 (or 32x16 transposed) f16 tile.
// p points to the start of this lane's row's K-segment (32 halves, 16B aligned).
// Layout (CDNA5 ISA 7.12.2): halves 0..7 = K kg*8..+7 ; halves 8..15 = K 16+kg*8..+7
DEV_INLINE v16h load_frag32(const _Float16* p, int kg) {
  union { v16h v; v8h h[2]; } f;
  f.h[0] = *(const v8h*)(p + kg * 8);
  f.h[1] = *(const v8h*)(p + 16 + kg * 8);
  return f.v;
}

DEV_INLINE float sigmoidf_(float x) { return 1.f / (1.f + expf(-x)); }

// ---------------------------------------------------------------------------
// x = [block_features | role_emb[role] | idx_emb[row%1024] | zero-pad to 64], f16
__global__ void build_x_kernel(const float* __restrict__ bf,
                               const int* __restrict__ role_id,
                               const float* __restrict__ role_emb,
                               const float* __restrict__ idx_emb,
                               _Float16* __restrict__ xpad) {
  unsigned row = blockIdx.x;
  unsigned c = threadIdx.x;            // 64 threads
  float v = 0.f;
  if (c < 16)       v = bf[row * 16u + c];
  else if (c < 32)  v = role_emb[(unsigned)role_id[row] * 16u + (c - 16)];
  else if (c < 40)  v = idx_emb[(row & 1023u) * 8u + (c - 32)];
  xpad[row * 64u + c] = (_Float16)v;
}

// ---------------------------------------------------------------------------
// dst[n*Kpad + k] = (k < K) ? (f16)src[k*Nc + n] : 0   (transpose + f32->f16)
__global__ void conv_transpose_f16(const float* __restrict__ src,
                                   _Float16* __restrict__ dst,
                                   int K, int Nc, int Kpad) {
  int id = blockIdx.x * blockDim.x + threadIdx.x;
  if (id >= Nc * Kpad) return;
  int n = id / Kpad, k = id % Kpad;
  dst[id] = (k < K) ? (_Float16)src[(unsigned)k * (unsigned)Nc + (unsigned)n]
                    : (_Float16)0.f;
}

__global__ void conv_f16_kernel(const float* __restrict__ src,
                                _Float16* __restrict__ dst, int total) {
  int id = blockIdx.x * blockDim.x + threadIdx.x;
  if (id < total) dst[id] = (_Float16)src[id];
}

// rel_p = rel_emb (6x128) @ msg_W1[2H:3H] (128x128)  -- tiny, plain VALU
__global__ void relp_kernel(const float* __restrict__ rel_emb,
                            const float* __restrict__ msg_W1,
                            float* __restrict__ relp) {
  unsigned r = blockIdx.x, c = threadIdx.x;  // 6 x 128
  float acc = 0.f;
  for (unsigned k = 0; k < HH; ++k)
    acc += rel_emb[r * HH + k] * msg_W1[(2u * HH + k) * HH + c];
  relp[r * HH + c] = acc;
}

// ---------------------------------------------------------------------------
// WMMA GEMM: C(M x Nc) = op(A_f16(M x K, row-major) @ Bt_f16(Nc x K)ᵀ + bias)
// One wave computes a 16x64 strip (A fragment reused across 4 WMMAs).
// grid = (M/16, Nc/64), block = 32.
template <bool RELU, bool WF32, bool WF16>
__global__ void gemm_wmma(const _Float16* __restrict__ A,
                          const _Float16* __restrict__ Bt,
                          const float* __restrict__ bias,
                          float* __restrict__ Cf32,
                          _Float16* __restrict__ Cf16,
                          int K) {
  unsigned mtile = blockIdx.x, ntile = blockIdx.y;
  unsigned Nc = gridDim.y * 64u;
  unsigned lane = threadIdx.x & 31;
  unsigned mn = lane & 15, kg = lane >> 4;
  const _Float16* arow = A + (mtile * 16u + mn) * (unsigned)K;
  const _Float16* bbase = Bt + (ntile * 64u + mn) * (unsigned)K;
  v8f acc[4] = {};
  for (int kb = 0; kb < K; kb += 32) {
    v16h a = load_frag32(arow + kb, kg);
#pragma unroll
    for (int nt = 0; nt < 4; ++nt) {
      v16h b = load_frag32(bbase + (unsigned)nt * 16u * (unsigned)K + kb, kg);
      acc[nt] = __builtin_amdgcn_wmma_f32_16x16x32_f16(false, a, false, b,
                                                       (short)0, acc[nt], false, false);
    }
  }
#pragma unroll
  for (int nt = 0; nt < 4; ++nt) {
    unsigned col = ntile * 64u + (unsigned)nt * 16u + mn;
    float bv = bias ? bias[col] : 0.f;
#pragma unroll
    for (int v = 0; v < 8; ++v) {
      unsigned row = mtile * 16u + (unsigned)v + 8u * kg;  // C/D layout: VGPR v -> M=v+8*kg
      float x = acc[nt][v] + bv;
      if (RELU) x = fmaxf(x, 0.f);
      if (WF32) Cf32[row * Nc + col] = x;
      if (WF16) Cf16[row * Nc + col] = (_Float16)x;
    }
  }
}

// ---------------------------------------------------------------------------
// Edge stage: 4 waves/block, 8 edge-tiles (of 16 edges) per wave.
// msg_W2ᵀ (32KB f16) staged once per block into LDS with CDNA5 async copies;
// per tile: build relu(pre) in LDS (f16), WMMA vs LDS-resident B, scatter-add
// (+msg_b2) into agg[edge_dst] with f32 atomics (u32 saddr-form addressing).
static constexpr int EDGE_TPW = 8;  // tiles per wave
__global__ void __launch_bounds__(128)
edge_msg_kernel(const int* __restrict__ es, const int* __restrict__ ed,
                const int* __restrict__ er, const float* __restrict__ ew,
                const float* __restrict__ src_p, const float* __restrict__ dst_p,
                const float* __restrict__ rel_p,
                const float* __restrict__ Ww,  const float* __restrict__ b1,
                const _Float16* __restrict__ W2t, const float* __restrict__ b2,
                float* __restrict__ agg) {
  __shared__ __align__(16) _Float16 sB[HH * HH];      // 32 KB: msg_W2t, N-major
  __shared__ __align__(16) _Float16 sA[4][16 * HH];   // 4 KB per wave
  __shared__ int sDst[4][16];

  unsigned tid = threadIdx.x;
  unsigned wv = tid >> 5, lane = tid & 31;

  // --- async-stage W2t into LDS: 2048 x 16B chunks across 128 threads ---
  {
    const char* gsrc = (const char*)W2t;
    unsigned lbase = (unsigned)(uintptr_t)&sB[0];
#pragma unroll
    for (int it = 0; it < 16; ++it) {
      unsigned chunk = (unsigned)it * 128u + tid;
      unsigned long long ga = (unsigned long long)(uintptr_t)(gsrc + chunk * 16u);
      unsigned la = lbase + chunk * 16u;
      asm volatile("global_load_async_to_lds_b128 %0, %1, off"
                   :: "v"(la), "v"(ga) : "memory");
    }
    asm volatile("s_wait_asynccnt 0x0" ::: "memory");
  }
  __syncthreads();

  float wwr[4], bb1[4];
#pragma unroll
  for (int j = 0; j < 4; ++j) {
    wwr[j] = Ww[lane + 32u * j];
    bb1[j] = b1[lane + 32u * j];
  }

  unsigned mn = lane & 15, kg = lane >> 4;
  _Float16* myA = &sA[wv][0];
  int* myDst = &sDst[wv][0];
  unsigned tile0 = (blockIdx.x * 4u + wv) * EDGE_TPW;

  for (int t = 0; t < EDGE_TPW; ++t) {
    unsigned e0 = (tile0 + (unsigned)t) * 16u;
    // build A tile: relu(src_p[s] + dst_p[d] + rel_p[r] + w*Ww + b1) as f16
    for (int i = 0; i < 16; ++i) {
      unsigned e = e0 + (unsigned)i;
      unsigned s = (unsigned)es[e], d = (unsigned)ed[e], r = (unsigned)er[e];
      float w = ew[e];
      if (lane == 0) myDst[i] = (int)d;
      unsigned so = s * (unsigned)HH, dofs = d * (unsigned)HH, ro = r * (unsigned)HH;
#pragma unroll
      for (int j = 0; j < 4; ++j) {
        unsigned c = lane + 32u * j;
        float p = src_p[so + c] + dst_p[dofs + c] + rel_p[ro + c] + w * wwr[j] + bb1[j];
        myA[(unsigned)i * HH + c] = (_Float16)fmaxf(p, 0.f);
      }
    }
    // per-wave LDS ops are in-order; no cross-wave sharing of myA

    v16h afr[4];
#pragma unroll
    for (int kb4 = 0; kb4 < 4; ++kb4)
      afr[kb4] = load_frag32(myA + mn * HH + kb4 * 32, kg);

#pragma unroll
    for (int nt = 0; nt < 8; ++nt) {
      const _Float16* brow = &sB[((unsigned)nt * 16u + mn) * HH];
      v8f acc = {};
#pragma unroll
      for (int kb4 = 0; kb4 < 4; ++kb4) {
        v16h b = load_frag32(brow + kb4 * 32, kg);
        acc = __builtin_amdgcn_wmma_f32_16x16x32_f16(false, afr[kb4], false, b,
                                                     (short)0, acc, false, false);
      }
      unsigned col = (unsigned)nt * 16u + mn;
      float bv = b2[col];
#pragma unroll
      for (int v = 0; v < 8; ++v) {
        unsigned m = (unsigned)v + 8u * kg;
        unsigned idx = (unsigned)myDst[m] * (unsigned)HH + col;  // u32: saddr+voffset form
        atomicAdd(agg + idx, acc[v] + bv);
      }
    }
  }
}

// ---------------------------------------------------------------------------
// Fused GRU gate + residual + LayerNorm + pooling accumulation. block = 128 (one row).
__global__ void gru_ln_kernel(const float* __restrict__ gi, const float* __restrict__ gh,
                              const float* __restrict__ hidden,
                              const float* __restrict__ gamma, const float* __restrict__ beta,
                              const float* __restrict__ bf,
                              float* __restrict__ out, float* __restrict__ pool) {
  unsigned row = blockIdx.x, c = threadIdx.x;
  const float* gir = gi + row * 3u * HH;
  const float* ghr = gh + row * 3u * HH;
  float h = hidden[row * HH + c];
  float r = sigmoidf_(gir[c] + ghr[c]);
  float z = sigmoidf_(gir[HH + c] + ghr[HH + c]);
  float cand = tanhf(gir[2 * HH + c] + r * ghr[2 * HH + c]);
  float upd = (1.f - z) * cand + z * h;
  float y = h + upd;

  __shared__ float red[HH];
  red[c] = y;
  __syncthreads();
  for (int s = 64; s > 0; s >>= 1) { if (c < s) red[c] += red[c + s]; __syncthreads(); }
  float mu = red[0] * (1.f / HH);
  __syncthreads();
  float dy = y - mu;
  red[c] = dy * dy;
  __syncthreads();
  for (int s = 64; s > 0; s >>= 1) { if (c < s) red[c] += red[c + s]; __syncthreads(); }
  float var = red[0] * (1.f / HH);

  float o = dy * rsqrtf(var + 1e-5f) * gamma[c] + beta[c];
  out[row * HH + c] = o;

  bool placed = bf[row * 16u + 10] > 0.5f;
  atomicAdd(&pool[c], o);
  atomicAdd(&pool[(placed ? HH : 2 * HH) + c], o);
  if (c == 0) atomicAdd(&pool[3 * HH + (placed ? 0 : 1)], 1.f);
}

// pooled(384) -> relu(@g_W1 + b1) -> @g_W2 + b2 -> 128. One block of 128 threads.
__global__ void graph_mlp_kernel(const float* __restrict__ pool,
                                 const float* __restrict__ g_W1, const float* __restrict__ g_b1,
                                 const float* __restrict__ g_W2, const float* __restrict__ g_b2,
                                 float* __restrict__ gout, float invN) {
  __shared__ float pooled[3 * HH];
  __shared__ float hbuf[HH];
  unsigned j = threadIdx.x;
  float cp = pool[3 * HH + 0], cu = pool[3 * HH + 1];
  pooled[j] = pool[j] * invN;
  pooled[HH + j]     = (cp > 0.f) ? pool[HH + j] / fmaxf(cp, 1.f) : 0.f;
  pooled[2 * HH + j] = (cu > 0.f) ? pool[2 * HH + j] / fmaxf(cu, 1.f) : 0.f;
  __syncthreads();
  float acc = g_b1[j];
  for (unsigned k = 0; k < 3 * HH; ++k) acc += pooled[k] * g_W1[k * HH + j];
  hbuf[j] = fmaxf(acc, 0.f);
  __syncthreads();
  float o = g_b2[j];
  for (unsigned k = 0; k < HH; ++k) o += hbuf[k] * g_W2[k * HH + j];
  gout[j] = o;
}

// ---------------------------------------------------------------------------
extern "C" void kernel_launch(void* const* d_in, const int* in_sizes, int n_in,
                              void* d_out, int out_size, void* d_ws, size_t ws_size,
                              hipStream_t stream) {
  (void)in_sizes; (void)n_in; (void)out_size; (void)ws_size;
  const float* bf       = (const float*)d_in[0];
  const int*   role_id  = (const int*)  d_in[1];
  const int*   esrc     = (const int*)  d_in[2];
  const int*   edst     = (const int*)  d_in[3];
  const int*   erel     = (const int*)  d_in[4];
  const float* ew       = (const float*)d_in[5];
  const float* role_emb = (const float*)d_in[6];
  const float* idx_emb  = (const float*)d_in[7];
  const float* in_W1    = (const float*)d_in[8];
  const float* in_b1    = (const float*)d_in[9];
  const float* in_W2    = (const float*)d_in[10];
  const float* in_b2    = (const float*)d_in[11];
  const float* rel_emb  = (const float*)d_in[12];
  const float* msg_W1   = (const float*)d_in[13];
  const float* msg_b1   = (const float*)d_in[14];
  const float* msg_W2   = (const float*)d_in[15];
  const float* msg_b2   = (const float*)d_in[16];
  const float* gwih     = (const float*)d_in[17];
  const float* gbih     = (const float*)d_in[18];
  const float* gwhh     = (const float*)d_in[19];
  const float* gbhh     = (const float*)d_in[20];
  const float* ln_g     = (const float*)d_in[21];
  const float* ln_b     = (const float*)d_in[22];
  const float* g_W1     = (const float*)d_in[23];
  const float* g_b1     = (const float*)d_in[24];
  const float* g_W2     = (const float*)d_in[25];
  const float* g_b2     = (const float*)d_in[26];
  float* out = (float*)d_out;

  char* ws = (char*)d_ws;
  size_t off = 0;
  auto alloc = [&](size_t bytes) {
    size_t o = off;
    off += (bytes + 255) & ~(size_t)255;
    return o;
  };
  _Float16* xpad   = (_Float16*)(ws + alloc((size_t)NN * 64 * 2));
  _Float16* h1_16  = (_Float16*)(ws + alloc((size_t)NN * HH * 2));
  _Float16* hid16  = (_Float16*)(ws + alloc((size_t)NN * HH * 2));
  float*    hid32  = (float*)   (ws + alloc((size_t)NN * HH * 4));
  float*    srcp   = (float*)   (ws + alloc((size_t)NN * HH * 4));
  float*    dstp   = (float*)   (ws + alloc((size_t)NN * HH * 4));
  float*    relp   = (float*)   (ws + alloc((size_t)6 * HH * 4));
  float*    agg    = (float*)   (ws + alloc((size_t)NN * HH * 4));
  _Float16* agg16  = (_Float16*)(ws + alloc((size_t)NN * HH * 2));
  float*    gi     = (float*)   (ws + alloc((size_t)NN * 3 * HH * 4));
  float*    gh     = (float*)   (ws + alloc((size_t)NN * 3 * HH * 4));
  _Float16* W1t    = (_Float16*)(ws + alloc((size_t)128 * 64 * 2));
  _Float16* W2t    = (_Float16*)(ws + alloc((size_t)128 * 128 * 2));
  _Float16* Wst    = (_Float16*)(ws + alloc((size_t)128 * 128 * 2));
  _Float16* Wdt    = (_Float16*)(ws + alloc((size_t)128 * 128 * 2));
  _Float16* mW2t   = (_Float16*)(ws + alloc((size_t)128 * 128 * 2));
  _Float16* gwih_t = (_Float16*)(ws + alloc((size_t)384 * 128 * 2));
  _Float16* gwhh_t = (_Float16*)(ws + alloc((size_t)384 * 128 * 2));
  float*    pool   = (float*)   (ws + alloc((size_t)512 * 4));

  // zero accumulators every call (ws is not re-poisoned between replays)
  hipMemsetAsync(agg, 0, (size_t)NN * HH * 4, stream);
  hipMemsetAsync(pool, 0, 512 * 4, stream);

  // input feature assembly + weight conversion (f32 -> f16, N-major transpose)
  build_x_kernel<<<NN, 64, 0, stream>>>(bf, role_id, role_emb, idx_emb, xpad);
  auto launch_tc = [&](const float* src, _Float16* dst, int K, int Nc, int Kpad) {
    int total = Nc * Kpad;
    conv_transpose_f16<<<(total + 255) / 256, 256, 0, stream>>>(src, dst, K, Nc, Kpad);
  };
  launch_tc(in_W1, W1t, 40, 128, 64);
  launch_tc(in_W2, W2t, 128, 128, 128);
  launch_tc(msg_W1, Wst, 128, 128, 128);
  launch_tc(msg_W1 + 128 * 128, Wdt, 128, 128, 128);
  launch_tc(msg_W2, mW2t, 128, 128, 128);
  launch_tc(gwih, gwih_t, 128, 384, 128);
  launch_tc(gwhh, gwhh_t, 128, 384, 128);
  relp_kernel<<<6, 128, 0, stream>>>(rel_emb, msg_W1, relp);

  // input MLP (WMMA): hidden = relu(relu(x@W1+b1)@W2+b2)
  gemm_wmma<true, false, true><<<dim3(NN / 16, 2), 32, 0, stream>>>(
      xpad, W1t, in_b1, nullptr, h1_16, 64);
  gemm_wmma<true, true, true><<<dim3(NN / 16, 2), 32, 0, stream>>>(
      h1_16, W2t, in_b2, hid32, hid16, 128);

  // node projections for edge messages (WMMA)
  gemm_wmma<false, true, false><<<dim3(NN / 16, 2), 32, 0, stream>>>(
      hid16, Wst, nullptr, srcp, nullptr, 128);
  gemm_wmma<false, true, false><<<dim3(NN / 16, 2), 32, 0, stream>>>(
      hid16, Wdt, nullptr, dstp, nullptr, 128);

  // dominant edge GEMM + segment-sum scatter (async-staged B in LDS)
  edge_msg_kernel<<<EE / (16 * 4 * EDGE_TPW), 128, 0, stream>>>(
      esrc, edst, erel, ew, srcp, dstp, relp,
      msg_W1 + 384 * 128, msg_b1, mW2t, msg_b2, agg);

  // GRU gate GEMMs (WMMA)
  conv_f16_kernel<<<(NN * HH + 255) / 256, 256, 0, stream>>>(agg, agg16, NN * HH);
  gemm_wmma<false, true, false><<<dim3(NN / 16, 6), 32, 0, stream>>>(
      agg16, gwih_t, gbih, gi, nullptr, 128);
  gemm_wmma<false, true, false><<<dim3(NN / 16, 6), 32, 0, stream>>>(
      hid16, gwhh_t, gbhh, gh, nullptr, 128);

  // GRU + residual + LayerNorm + pooling sums
  gru_ln_kernel<<<NN, 128, 0, stream>>>(gi, gh, hid32, ln_g, ln_b, bf, out, pool);

  // graph-level MLP tail
  graph_mlp_kernel<<<1, 128, 0, stream>>>(pool, g_W1, g_b1, g_W2, g_b2,
                                          out + (size_t)NN * HH, 1.f / (float)NN);
}
